// Block_77318001263203
// MI455X (gfx1250) — matrix operand order repaired and checked by reading, not compiled, not run on hardware
//
#include <hip/hip_runtime.h>
#include <hip/hip_bf16.h>

typedef unsigned short u16;
typedef __attribute__((ext_vector_type(16))) __bf16 v16bf;
typedef __attribute__((ext_vector_type(8)))  float  v8f;

// ---- problem constants --------------------------------------------------
#define D      1024
#define FDIM   4096
#define SEQL   2048
#define BATCH  2
#define NHEAD  16
#define HSZ    64
#define ROWS   (BATCH * SEQL)   // 4096
#define LN_EPS 1e-5f

// ---- helpers ------------------------------------------------------------
__device__ __forceinline__ u16 f2bf(float f) {
  unsigned u = __builtin_bit_cast(unsigned, f);
  return (u16)((u + 0x7fffu + ((u >> 16) & 1u)) >> 16);
}

union Frag {
  uint4 q[2];
  v16bf v;
};

__device__ __forceinline__ v8f wmma_bf16(v16bf a, v16bf b, v8f c) {
  return __builtin_amdgcn_wmma_f32_16x16x32_bf16(false, a, false, b, (short)0, c,
                                                 false, false);
}

__device__ __forceinline__ v8f vzero8() {
  v8f z;
#pragma unroll
  for (int i = 0; i < 8; ++i) z[i] = 0.0f;
  return z;
}

// ---- elementwise conversion kernels ------------------------------------
__global__ __launch_bounds__(256) void cvt_f32_bf16(const float* __restrict__ s,
                                                    u16* __restrict__ d, int n) {
  int i = blockIdx.x * 256 + threadIdx.x;
  if (i < n) d[i] = f2bf(s[i]);
}

// wq/wk/wv: [H, D, hs] -> bf16 [D, H*hs]
__global__ __launch_bounds__(256) void remap_qkv(const float* __restrict__ s,
                                                 u16* __restrict__ d) {
  int i = blockIdx.x * 256 + threadIdx.x;  // i over D*D
  int dd = i >> 10;           // row (input feature)
  int c  = i & 1023;          // column = h*64 + k
  int h  = c >> 6;
  int kk = c & 63;
  d[i] = f2bf(s[(size_t)h * (D * HSZ) + (size_t)dd * HSZ + kk]);
}

// ---- tiled bf16 WMMA GEMM ----------------------------------------------
// C[M,N] = act(A[M,K] @ B[K,N] + bias)
// block tile 256x64, K-step 32, 256 threads = 8 waves, each wave: 32x64 strip
// (2 A-fragments x 4 B-fragments = 8 WMMA per K-step per wave).
// A tiles staged with CDNA5 async global->LDS copies (ASYNCcnt), double
// buffered; B tiles staged via register transpose into LDS.
// outMode: 0 = f32, 1 = bf16, 2 = bf16 transposed-per-head (for V):
//          dst[((b*16+h)*64+n)*SEQL + s]
#define LDSS 40   // padded LDS row stride (u16 units); 80B, 16B-aligned
#define BM   256
#define A_WORDS (BM * LDSS)          // one A buffer, u16 units
#define B_WORDS (64 * LDSS)          // one B buffer, u16 units

__global__ __launch_bounds__(256) void gemm_bf16(
    const u16* __restrict__ A, const u16* __restrict__ B,
    const float* __restrict__ bias, void* __restrict__ Cout, int M, int N,
    int K, int outMode, int relu) {
  // single master LDS array so async-copy LDS offsets are known (segment base 0)
  __shared__ __align__(16) u16 SMEM[2 * A_WORDS + 2 * B_WORDS];
#define AS(buf, idx) SMEM[(buf) * A_WORDS + (idx)]
#define BS(buf, idx) SMEM[2 * A_WORDS + (buf) * B_WORDS + (idx)]

  const int tid  = threadIdx.x;
  const int wave = tid >> 5;
  const int lane = tid & 31;
  const int half = lane >> 4;
  const int l16  = lane & 15;
  const int mbase = blockIdx.y * BM;
  const int nbase = blockIdx.x * 64;

  v8f acc[2][4];
#pragma unroll
  for (int i = 0; i < 2; ++i)
#pragma unroll
    for (int t = 0; t < 4; ++t) acc[i][t] = vzero8();

  // stage A tile 256x32 via async global->LDS b128 copies (1 row / thread)
  auto stageA = [&](int buf, int k0) {
    unsigned long long ga =
        (unsigned long long)(uintptr_t)(A + (size_t)(mbase + tid) * K + k0);
    unsigned lo = (unsigned)((buf * A_WORDS + tid * LDSS) * 2);  // LDS byte off
    asm volatile("global_load_async_to_lds_b128 %0, %1, off" ::"v"(lo), "v"(ga)
                 : "memory");
    asm volatile("global_load_async_to_lds_b128 %0, %1, off offset:16" ::"v"(lo),
                 "v"(ga)
                 : "memory");
    asm volatile("global_load_async_to_lds_b128 %0, %1, off offset:32" ::"v"(lo),
                 "v"(ga)
                 : "memory");
    asm volatile("global_load_async_to_lds_b128 %0, %1, off offset:48" ::"v"(lo),
                 "v"(ga)
                 : "memory");
  };
  // stage B tile 32x64 transposed into Bs[n][k]
  auto stageB = [&](int buf, int k0) {
    int kk = tid >> 3, n0 = (tid & 7) * 8;
    const u16* src = B + (size_t)(k0 + kk) * N + nbase + n0;
    union { uint4 u; u16 s[8]; } t;
    t.u = *(const uint4*)(src);
#pragma unroll
    for (int i = 0; i < 8; ++i) BS(buf, (n0 + i) * LDSS + kk) = t.s[i];
  };

  stageA(0, 0);
  stageB(0, 0);

  const int nk = K >> 5;
  for (int it = 0; it < nk; ++it) {
    const int cur = it & 1, nxt = cur ^ 1;
    // our async copies (and ds stores, via compiler) must land before barrier
    asm volatile("s_wait_asynccnt 0x0" ::: "memory");
    __syncthreads();
    if (it + 1 < nk) {  // prefetch next tile while computing this one
      stageA(nxt, (it + 1) * 32);
      stageB(nxt, (it + 1) * 32);
    }

    // A fragments: rows wave*32 + {l16, 16+l16};
    // ISA A layout: K = {half*8..+7, 16+half*8..+7}
    Frag af0, af1;
    const int ar = wave * 32 + l16;
    af0.q[0] = *(const uint4*)&AS(cur, ar * LDSS + half * 8);
    af0.q[1] = *(const uint4*)&AS(cur, ar * LDSS + 16 + half * 8);
    af1.q[0] = *(const uint4*)&AS(cur, (ar + 16) * LDSS + half * 8);
    af1.q[1] = *(const uint4*)&AS(cur, (ar + 16) * LDSS + 16 + half * 8);

#pragma unroll
    for (int nt = 0; nt < 4; ++nt) {
      Frag bf;
      const int brow = nt * 16 + l16;  // Bs is [n][k]; K = {half*8..+7, 16+half*8..+7}
      bf.q[0] = *(const uint4*)&BS(cur, brow * LDSS + half * 8);
      bf.q[1] = *(const uint4*)&BS(cur, brow * LDSS + 16 + half * 8);
      acc[0][nt] = wmma_bf16(af0.v, bf.v, acc[0][nt]);
      acc[1][nt] = wmma_bf16(af1.v, bf.v, acc[1][nt]);
    }
  }

  // epilogue: acc[i][nt] element r = (row mbase+wave*32+i*16+r+half*8,
  //                                   col nbase+nt*16+l16)
#pragma unroll
  for (int nt = 0; nt < 4; ++nt) {
    const int col = nbase + nt * 16 + l16;
    const float bs = bias ? bias[col] : 0.0f;
#pragma unroll
    for (int i = 0; i < 2; ++i) {
#pragma unroll
      for (int r = 0; r < 8; ++r) {
        const int row = mbase + wave * 32 + i * 16 + r + half * 8;
        float v = acc[i][nt][r] + bs;
        if (relu) v = fmaxf(v, 0.0f);
        if (outMode == 0) {
          ((float*)Cout)[(size_t)row * N + col] = v;
        } else if (outMode == 1) {
          ((u16*)Cout)[(size_t)row * N + col] = f2bf(v);
        } else {  // V transpose: [B,H,hs,S]
          int h = col >> 6, n = col & 63;
          int bsel = row >> 11, s = row & (SEQL - 1);
          ((u16*)Cout)[(((size_t)bsel * NHEAD + h) * HSZ + n) * SEQL + s] =
              f2bf(v);
        }
      }
    }
  }
#undef AS
#undef BS
}

// ---- flash attention (causal) ------------------------------------------
// q,k: bf16 [B*S, H*hs] ; vt: bf16 [B,H,hs,S] ; out attn: bf16 [B*S, H*hs]
// 1 wave per 16-query block; 4 waves per workgroup (independent tasks).
__global__ __launch_bounds__(128) void attn_kernel(const u16* __restrict__ qg,
                                                   const u16* __restrict__ kg,
                                                   const u16* __restrict__ vtg,
                                                   u16* __restrict__ attn) {
  __shared__ __align__(16) u16 P[4 * 16 * LDSS];

  const int wave = threadIdx.x >> 5;
  const int lane = threadIdx.x & 31;
  const int half = lane >> 4;
  const int l16  = lane & 15;

  const int task = blockIdx.x * 4 + wave;   // B*H*(S/16) = 4096 tasks
  const int qb = task & 127;
  const int bh = task >> 7;
  const int b  = bh >> 4;
  const int h  = bh & 15;

  u16* Pw = &P[wave * 16 * LDSS];

  // Q A-fragments for the two K=32 chunks of hs=64
  Frag qa[2];
  {
    const u16* qrow = qg + ((size_t)(b * SEQL + qb * 16 + l16)) * D + h * HSZ;
#pragma unroll
    for (int c = 0; c < 2; ++c) {
      qa[c].q[0] = *(const uint4*)(qrow + c * 32 + half * 8);
      qa[c].q[1] = *(const uint4*)(qrow + c * 32 + 16 + half * 8);
    }
  }

  float mrow[8], lrow[8];
  v8f o[4];
#pragma unroll
  for (int r = 0; r < 8; ++r) { mrow[r] = -3.0e38f; lrow[r] = 0.0f; }
#pragma unroll
  for (int t = 0; t < 4; ++t) o[t] = vzero8();

  const int kend = qb * 16 + 16;
  for (int kt = 0; kt < kend; kt += 32) {
    v8f s0 = vzero8(), s1 = vzero8();
    // scores = Q @ K^T ; B-operand b[j] = K[key(n)][c*32 + (j/8)*16 + half*8 + j%8]
#pragma unroll
    for (int c = 0; c < 2; ++c) {
      Frag kb0, kb1;
      const u16* kr0 =
          kg + ((size_t)(b * SEQL + kt + l16)) * D + h * HSZ + c * 32 + half * 8;
      kb0.q[0] = *(const uint4*)(kr0);
      kb0.q[1] = *(const uint4*)(kr0 + 16);
      s0 = wmma_bf16(qa[c].v, kb0.v, s0);
      const u16* kr1 = kg + ((size_t)(b * SEQL + kt + 16 + l16)) * D + h * HSZ +
                       c * 32 + half * 8;
      kb1.q[0] = *(const uint4*)(kr1);
      kb1.q[1] = *(const uint4*)(kr1 + 16);
      s1 = wmma_bf16(qa[c].v, kb1.v, s1);
    }

    // online softmax; row r lives across the 16-lane group (shfl masks 1..8)
#pragma unroll
    for (int r = 0; r < 8; ++r) {
      const int qrow_i = qb * 16 + r + half * 8;
      float x0 = s0[r] * 0.125f;
      if (kt + l16 > qrow_i) x0 = -3.0e38f;
      float x1 = s1[r] * 0.125f;
      if (kt + 16 + l16 > qrow_i) x1 = -3.0e38f;
      float rm = fmaxf(x0, x1);
#pragma unroll
      for (int m = 8; m; m >>= 1) rm = fmaxf(rm, __shfl_xor(rm, m, 32));
      const float mo = mrow[r];
      const float mn = fmaxf(mo, rm);
      const float alpha = __expf(mo - mn);
      mrow[r] = mn;
      x0 = __expf(x0 - mn);
      x1 = __expf(x1 - mn);
      float rs = x0 + x1;
#pragma unroll
      for (int m = 8; m; m >>= 1) rs += __shfl_xor(rs, m, 32);
      lrow[r] = lrow[r] * alpha + rs;
#pragma unroll
      for (int t = 0; t < 4; ++t) o[t][r] *= alpha;
      // P -> LDS (C-tile layout -> row-major bf16 16x32)
      Pw[(r + half * 8) * LDSS + l16]      = f2bf(x0);
      Pw[(r + half * 8) * LDSS + 16 + l16] = f2bf(x1);
    }
    // same-wave LDS RAW: DS is in-order, but make it explicit on CDNA5
    __asm__ volatile("s_wait_dscnt 0x0" ::: "memory");

    // P A-fragment (16x32)
    Frag pa;
    pa.q[0] = *(const uint4*)&Pw[l16 * LDSS + half * 8];
    pa.q[1] = *(const uint4*)&Pw[l16 * LDSS + 16 + half * 8];

    // O += P @ V ; b[j] = V[kt + (j/8)*16 + half*8 + j%8][nt*16 + l16] via transposed vt
#pragma unroll
    for (int nt = 0; nt < 4; ++nt) {
      Frag vb;
      const u16* vr = vtg +
                      (((size_t)(b * NHEAD + h) * HSZ + nt * 16 + l16)) * SEQL +
                      kt + half * 8;
      vb.q[0] = *(const uint4*)(vr);
      vb.q[1] = *(const uint4*)(vr + 16);
      o[nt] = wmma_bf16(pa.v, vb.v, o[nt]);
    }
  }

  // normalize + store bf16
#pragma unroll
  for (int r = 0; r < 8; ++r) {
    const float inv = 1.0f / lrow[r];
    const size_t row = (size_t)(b * SEQL + qb * 16 + r + half * 8);
#pragma unroll
    for (int nt = 0; nt < 4; ++nt) {
      attn[row * D + h * HSZ + nt * 16 + l16] = f2bf(o[nt][r] * inv);
    }
  }
}

// ---- fused residual + LayerNorm ----------------------------------------
__device__ __forceinline__ float block_reduce(float val, float* sh) {
#pragma unroll
  for (int m = 16; m; m >>= 1) val += __shfl_xor(val, m, 32);
  const int w = threadIdx.x >> 5, lane = threadIdx.x & 31;
  __syncthreads();
  if (lane == 0) sh[w] = val;
  __syncthreads();
  float t = sh[0];
#pragma unroll
  for (int i = 1; i < 8; ++i) t += sh[i];
  return t;
}

__global__ __launch_bounds__(256) void ln_kernel(
    const float* __restrict__ xr, const float* __restrict__ yr,
    const float* __restrict__ g, const float* __restrict__ be,
    float* __restrict__ of, u16* __restrict__ ob) {
  __shared__ float sh[8];
  const int row = blockIdx.x;
  const int tid = threadIdx.x;
  const float* xp = xr + (size_t)row * D;
  const float* yp = yr + (size_t)row * D;
  float v[4];
  float s = 0.0f;
#pragma unroll
  for (int i = 0; i < 4; ++i) {
    const int c = tid + i * 256;
    v[i] = xp[c] + yp[c];
    s += v[i];
  }
  s = block_reduce(s, sh);
  const float mean = s * (1.0f / D);
  float s2 = 0.0f;
#pragma unroll
  for (int i = 0; i < 4; ++i) {
    const float d = v[i] - mean;
    s2 += d * d;
  }
  s2 = block_reduce(s2, sh);
  const float rstd = rsqrtf(s2 * (1.0f / D) + LN_EPS);
#pragma unroll
  for (int i = 0; i < 4; ++i) {
    const int c = tid + i * 256;
    const float o = (v[i] - mean) * rstd * g[c] + be[c];
    of[(size_t)row * D + c] = o;
    if (ob) ob[(size_t)row * D + c] = f2bf(o);
  }
}

// ---- driver -------------------------------------------------------------
extern "C" void kernel_launch(void* const* d_in, const int* in_sizes, int n_in,
                              void* d_out, int out_size, void* d_ws,
                              size_t ws_size, hipStream_t stream) {
  const float* x      = (const float*)d_in[0];
  const float* wq     = (const float*)d_in[1];
  const float* wk     = (const float*)d_in[2];
  const float* wv     = (const float*)d_in[3];
  const float* w_proj = (const float*)d_in[4];
  const float* b_proj = (const float*)d_in[5];
  const float* w1     = (const float*)d_in[6];
  const float* b1     = (const float*)d_in[7];
  const float* w2     = (const float*)d_in[8];
  const float* b2     = (const float*)d_in[9];
  const float* g1     = (const float*)d_in[10];
  const float* be1    = (const float*)d_in[11];
  const float* g2     = (const float*)d_in[12];
  const float* be2    = (const float*)d_in[13];
  float* out = (float*)d_out;

  const size_t MB = 1u << 20;
  char* w = (char*)d_ws;
  // workspace layout (byte offsets), total 104 MB:
  //   0..8    xb (bf16 x; reused later as attn)
  //   8..14   wqb, wkb, wvb (2 MB each)
  //   14..16  wpb
  //   16..24  w1b
  //   24..32  w2b
  //   32..56  q, k, vt (8 MB each); h1 (32 MB) overlays 32..64 after attention
  //   64..80  y (f32; proj out, later mlp2 out)
  //   80..96  xn1 (f32)
  //   96..104 xn1b (bf16)
  u16* xb   = (u16*)(w + 0);
  u16* wqb  = (u16*)(w + 8 * MB);
  u16* wkb  = (u16*)(w + 10 * MB);
  u16* wvb  = (u16*)(w + 12 * MB);
  u16* wpb  = (u16*)(w + 14 * MB);
  u16* w1b  = (u16*)(w + 16 * MB);
  u16* w2b  = (u16*)(w + 24 * MB);
  u16* qb_  = (u16*)(w + 32 * MB);
  u16* kb_  = (u16*)(w + 40 * MB);
  u16* vtb  = (u16*)(w + 48 * MB);
  u16* h1   = (u16*)(w + 32 * MB);
  float* y  = (float*)(w + 64 * MB);
  float* xn1  = (float*)(w + 80 * MB);
  u16*   xn1b = (u16*)(w + 96 * MB);
  u16* attnb = xb;  // reuse after V GEMM

  // 1) conversions / weight remaps
  cvt_f32_bf16<<<(ROWS * D) / 256, 256, 0, stream>>>(x, xb, ROWS * D);
  remap_qkv<<<(D * D) / 256, 256, 0, stream>>>(wq, wqb);
  remap_qkv<<<(D * D) / 256, 256, 0, stream>>>(wk, wkb);
  remap_qkv<<<(D * D) / 256, 256, 0, stream>>>(wv, wvb);
  cvt_f32_bf16<<<(D * D) / 256, 256, 0, stream>>>(w_proj, wpb, D * D);
  cvt_f32_bf16<<<(D * FDIM) / 256, 256, 0, stream>>>(w1, w1b, D * FDIM);
  cvt_f32_bf16<<<(FDIM * D) / 256, 256, 0, stream>>>(w2, w2b, FDIM * D);

  // 2) QKV projections (bf16 WMMA GEMMs)
  dim3 gDD(D / 64, ROWS / BM);
  gemm_bf16<<<gDD, 256, 0, stream>>>(xb, wqb, nullptr, qb_, ROWS, D, D, 1, 0);
  gemm_bf16<<<gDD, 256, 0, stream>>>(xb, wkb, nullptr, kb_, ROWS, D, D, 1, 0);
  gemm_bf16<<<gDD, 256, 0, stream>>>(xb, wvb, nullptr, vtb, ROWS, D, D, 2, 0);

  // 3) causal flash attention
  attn_kernel<<<(BATCH * NHEAD * (SEQL / 16)) / 4, 128, 0, stream>>>(qb_, kb_,
                                                                     vtb, attnb);

  // 4) output projection (+bias), then residual + LN1
  gemm_bf16<<<gDD, 256, 0, stream>>>(attnb, wpb, b_proj, y, ROWS, D, D, 0, 0);
  ln_kernel<<<ROWS, 256, 0, stream>>>(x, y, g1, be1, xn1, xn1b);

  // 5) MLP
  dim3 gDF(FDIM / 64, ROWS / BM);
  gemm_bf16<<<gDF, 256, 0, stream>>>(xn1b, w1b, b1, h1, ROWS, FDIM, D, 1, 1);
  gemm_bf16<<<gDD, 256, 0, stream>>>(h1, w2b, b2, y, ROWS, D, FDIM, 0, 0);

  // 6) residual + LN2 -> final output (f32)
  ln_kernel<<<ROWS, 256, 0, stream>>>(xn1, y, g2, be2, out, nullptr);
}
